// SpatiotemporalTimelyGPT_31009663877439
// MI455X (gfx1250) — matrix operand order repaired and checked by reading, not compiled, not run on hardware
//
#include <hip/hip_runtime.h>
#include <hip/hip_bf16.h>
#include <math.h>

// ---------------------------------------------------------------------------
// Spatiotemporal TimelyGPT forward for MI455X (gfx1250), wave32 + WMMA.
//
// Algorithm: retention is causal with exponential decay, so the reference's
// "recompute whole buffer each step" loop is equivalent to incremental
// decoding with a per-(b,h) 64x64 recurrent state:  S = gamma*S + k^T v,
// ret = (q/8) @ S.  All heavy math becomes [256 x K] x [K x N] GEMMs done
// with v_wmma_f32_16x16x32_f16 (f16 in, f32 accumulate).
//
// Input flattening assumption (setup_inputs() dict insertion order, nested
// dicts/lists depth-first):
//  0 x_genes  1 x_coords
//  2 enc_w1 3 enc_b1 4 enc_bn1_g 5 enc_bn1_b 6 enc_w2 7 enc_b2
//  8 enc_bn2_g 9 enc_bn2_b 10 mu_w 11 mu_b 12 lv_w 13 lv_b
// 14 sp_w1 15 sp_b1 16 sp_ln_g 17 sp_ln_b 18 sp_w2 19 sp_b2
// 20+14*l+{0:ln1_g 1:ln1_b 2:wq 3:wk 4:wv 5:wg 6:wo 7:gn_g 8:ln2_g 9:ln2_b
//          10:ffn_w1 11:ffn_b1 12:ffn_w2 13:ffn_b2}   (l = 0..3)
// 76 ln_f_g 77 ln_f_b 78 dec_w1 79 dec_b1 80 dec_w2 81 dec_b2 82 dec_w3 83 dec_b3
// ---------------------------------------------------------------------------

typedef __attribute__((ext_vector_type(16))) _Float16 v16h;
typedef __attribute__((ext_vector_type(8)))  _Float16 v8h;
typedef __attribute__((ext_vector_type(8)))  float    v8f;

#define B_   256
#define D_   512
#define H_   8
#define DH_  64
#define T_   32
#define NG_  2000
#define FFN_ 1024

// ---------------------------------------------------------------------------
// Fragment loaders (wave32). A/B use this code's contiguous k order (not the ISA table order); C/D per ISA 7.12.2.
// A (16x32 f16, row-major, ld=K):  lane row = lane&15;
//   element i -> k = i + ((lane>=16)?16:0)
// B from transposed weights Wt[N][Kpad]: lane col = lane&15;
//   element i -> k = i + ((lane>=16)?16:0)   (16 contiguous halves)
// C/D (16x16 f32): lane col = lane&15; vgpr r -> row r + ((lane>=16)?8:0)
// ---------------------------------------------------------------------------
__device__ __forceinline__ v16h load_a_frag(const _Float16* __restrict__ A,
                                            size_t rowOff, int kpos) {
  const _Float16* p = A + rowOff + kpos;
  v16h a;
#pragma unroll
  for (int q = 0; q < 8; ++q) a[q] = p[q];           // k 0..7
#pragma unroll
  for (int q = 0; q < 8; ++q) a[8 + q] = p[8 + q];  // k 8..15
  return a;
}

__device__ __forceinline__ v16h load_b_frag(const _Float16* __restrict__ Wt,
                                            size_t colOff, int kpos) {
  const v8h* p = reinterpret_cast<const v8h*>(Wt + colOff + kpos);
  v8h lo = p[0], hi = p[1];
  v16h b;
#pragma unroll
  for (int q = 0; q < 8; ++q) { b[q] = lo[q]; b[8 + q] = hi[q]; }
  return b;
}

__device__ __forceinline__ float gelu_tanh(float x) {
  return 0.5f * x * (1.0f + tanhf(0.7978845608f * (x + 0.044715f * x * x * x)));
}

// ---------------------------------------------------------------------------
// GEMM: C[M,N] = act( A[M,K] * Wt[N,K]^T  (+bias)(*scale*sc)(+bias2)(+resid) )
// One wave -> 16 x (16*NT) tile. Block = (32,4) = 4 M-tiles.
// ---------------------------------------------------------------------------
template <int NT>
__global__ void gemm_wmma(const _Float16* __restrict__ A,
                          const _Float16* __restrict__ Wt,
                          const float* __restrict__ bias,
                          const float* __restrict__ scale, float sconst,
                          const float* __restrict__ bias2,
                          const float* __restrict__ resid,
                          float* __restrict__ o32, int ldc,
                          _Float16* __restrict__ o16,
                          int M, int N, int K, int act) {
  const int lane = threadIdx.x;
  const int mt = blockIdx.y * 4 + threadIdx.y;
  const int ntBase = blockIdx.x * NT;
  const int l15 = lane & 15;
  const int row = mt * 16 + l15;
  const size_t rowOff = (size_t)row * K;
  const int aLaneK = (lane >> 4) << 4;   // +16 for upper half-wave
  const int bLaneK = (lane >> 4) << 4;   // +16 for upper half-wave

  v8f acc[NT];
#pragma unroll
  for (int j = 0; j < NT; ++j) acc[j] = (v8f){};

  for (int kb = 0; kb < K; kb += 32) {
    v16h a = load_a_frag(A, rowOff, kb + aLaneK);
#pragma unroll
    for (int j = 0; j < NT; ++j) {
      size_t colOff = (size_t)((ntBase + j) * 16 + l15) * K;
      v16h b = load_b_frag(Wt, colOff, kb + bLaneK);
      acc[j] = __builtin_amdgcn_wmma_f32_16x16x32_f16(
          false, a, false, b, (short)0, acc[j], false, false);
    }
  }

  const int mOff = (lane >> 4) << 3;
#pragma unroll
  for (int j = 0; j < NT; ++j) {
    const int n = (ntBase + j) * 16 + l15;
#pragma unroll
    for (int r = 0; r < 8; ++r) {
      const int m = mt * 16 + r + mOff;
      float t = acc[j][r];
      if (bias)  t += bias[n];
      if (scale) t *= scale[n] * sconst;
      if (bias2) t += bias2[n];
      if (resid) t += resid[(size_t)m * N + n];
      if (act == 1)      t = fmaxf(t, 0.0f);
      else if (act == 2) t = gelu_tanh(t);
      if (o32) o32[(size_t)m * ldc + n] = t;
      if (o16) o16[(size_t)m * N + n] = (_Float16)t;
    }
  }
}

// ---------------------------------------------------------------------------
// Row LayerNorm (D = 128 or 512), optional relu, f16 + up to two f32 outputs.
// ---------------------------------------------------------------------------
__global__ void ln_kernel(const float* __restrict__ X, const float* __restrict__ g,
                          const float* __restrict__ bt, _Float16* __restrict__ o16,
                          float* __restrict__ o32a, int lda,
                          float* __restrict__ o32b, int ldb, int D, int act) {
  __shared__ float s1[128], s2[128];
  const int tid = threadIdx.x, row = blockIdx.x;
  const float* x = X + (size_t)row * D;
  const int per = D >> 7;
  float vals[4];
  float a0 = 0.f, a1 = 0.f;
  for (int i = 0; i < per; ++i) {
    float v = x[tid + (i << 7)];
    vals[i] = v; a0 += v; a1 += v * v;
  }
  s1[tid] = a0; s2[tid] = a1; __syncthreads();
  for (int s = 64; s > 0; s >>= 1) {
    if (tid < s) { s1[tid] += s1[tid + s]; s2[tid] += s2[tid + s]; }
    __syncthreads();
  }
  const float inv = 1.0f / (float)D;
  const float mean = s1[0] * inv;
  const float var = s2[0] * inv - mean * mean;
  const float rstd = rsqrtf(var + 1e-5f);
  for (int i = 0; i < per; ++i) {
    const int c = tid + (i << 7);
    float y = (vals[i] - mean) * rstd * g[c] + bt[c];
    if (act == 1) y = fmaxf(y, 0.0f);
    if (o16)  o16[(size_t)row * D + c] = (_Float16)y;
    if (o32a) o32a[(size_t)row * lda + c] = y;
    if (o32b) o32b[(size_t)row * ldb + c] = y;
  }
}

// ---------------------------------------------------------------------------
// Recurrent retention step for one token. Block = 64 threads = one (b,h).
// S[b,h] is 64x64 f32 in ws. qkvg layout per row: [q|k|v|g], each 512.
// Fuses group-norm (over DH) * gn_g and SiLU gate; emits f16 for wo GEMM.
// ---------------------------------------------------------------------------
__global__ void retention_kernel(const float* __restrict__ qkvg,
                                 float* __restrict__ S,
                                 const float* __restrict__ gn_g,
                                 _Float16* __restrict__ o16) {
  const int bh = blockIdx.x;
  const int b = bh >> 3, h = bh & 7;
  const int d2 = threadIdx.x;
  const float gamma = 1.0f - exp2f(-5.0f - (float)h);
  const float* q = qkvg + (size_t)b * 2048 + h * 64;
  const float* k = q + 512;
  const float* v = q + 1024;
  const float* gt = q + 1536;

  __shared__ float ksh[64], qsh[64], r1[64], r2[64];
  ksh[d2] = k[d2];
  qsh[d2] = q[d2] * 0.125f;           // 1/sqrt(DH)
  const float vv = v[d2];
  __syncthreads();

  float* Scol = S + (size_t)bh * 4096 + d2;  // S[d1][d2], coalesced over d2
  float ret = 0.f;
  for (int d1 = 0; d1 < 64; ++d1) {
    float s = gamma * Scol[d1 * 64] + ksh[d1] * vv;
    Scol[d1 * 64] = s;
    ret += qsh[d1] * s;
  }

  r1[d2] = ret; r2[d2] = ret * ret; __syncthreads();
  for (int s = 32; s > 0; s >>= 1) {
    if (d2 < s) { r1[d2] += r1[d2 + s]; r2[d2] += r2[d2 + s]; }
    __syncthreads();
  }
  const float mean = r1[0] * (1.0f / 64.0f);
  const float var = r2[0] * (1.0f / 64.0f) - mean * mean;
  const float y = (ret - mean) * rsqrtf(var + 1e-5f) * gn_g[h * 64 + d2];
  const float gv = gt[d2];
  const float silu = gv / (1.0f + expf(-gv));
  o16[(size_t)b * 512 + h * 64 + d2] = (_Float16)(y * silu);
}

// ---------------------------------------------------------------------------
// Small helper kernels.
// ---------------------------------------------------------------------------
__global__ void convert_wt(const float* __restrict__ W, _Float16* __restrict__ Wt,
                           int K, int N, int Kpad) {
  const size_t i = (size_t)blockIdx.x * 256 + threadIdx.x;
  const size_t total = (size_t)N * Kpad;
  if (i >= total) return;
  const int k = (int)(i % Kpad);
  const int n = (int)(i / Kpad);
  const float v = (k < K) ? W[(size_t)k * N + n] : 0.0f;
  Wt[i] = (_Float16)v;
}

__global__ void build_enc(const float* __restrict__ genes,
                          const float* __restrict__ coords,
                          _Float16* __restrict__ enc16) {
  const size_t i = (size_t)blockIdx.x * 256 + threadIdx.x;
  const size_t total = (size_t)B_ * 2016;
  if (i >= total) return;
  const int c = (int)(i % 2016);
  const int b = (int)(i / 2016);
  float v = 0.f;
  if (c < 2000) v = genes[(size_t)b * 2000 + c];
  else if (c < 2002) v = coords[b * 2 + (c - 2000)];
  enc16[i] = (_Float16)v;
}

__global__ void build_coords(const float* __restrict__ coords,
                             _Float16* __restrict__ co16) {
  const int i = blockIdx.x * 256 + threadIdx.x;
  if (i >= B_ * 32) return;
  const int c = i & 31, b = i >> 5;
  co16[i] = (_Float16)((c < 2) ? coords[b * 2 + c] : 0.0f);
}

__global__ void add2(const float* __restrict__ a, const float* __restrict__ b,
                     float* __restrict__ o, int n) {
  const int i = blockIdx.x * 256 + threadIdx.x;
  if (i < n) o[i] = a[i] + b[i];
}

// ---------------------------------------------------------------------------
// Host orchestration.
// ---------------------------------------------------------------------------
extern "C" void kernel_launch(void* const* d_in, const int* in_sizes, int n_in,
                              void* d_out, int out_size, void* d_ws, size_t ws_size,
                              hipStream_t stream) {
  (void)in_sizes; (void)n_in; (void)out_size; (void)ws_size;

  const float* x_genes  = (const float*)d_in[0];
  const float* x_coords = (const float*)d_in[1];
  const float* enc_w1 = (const float*)d_in[2];
  const float* enc_b1 = (const float*)d_in[3];
  const float* bn1_g  = (const float*)d_in[4];
  const float* bn1_b  = (const float*)d_in[5];
  const float* enc_w2 = (const float*)d_in[6];
  const float* enc_b2 = (const float*)d_in[7];
  const float* bn2_g  = (const float*)d_in[8];
  const float* bn2_b  = (const float*)d_in[9];
  const float* mu_w   = (const float*)d_in[10];
  const float* mu_b   = (const float*)d_in[11];
  const float* lv_w   = (const float*)d_in[12];
  const float* lv_b   = (const float*)d_in[13];
  const float* sp_w1  = (const float*)d_in[14];
  const float* sp_b1  = (const float*)d_in[15];
  const float* sp_ln_g = (const float*)d_in[16];
  const float* sp_ln_b = (const float*)d_in[17];
  const float* sp_w2  = (const float*)d_in[18];
  const float* sp_b2  = (const float*)d_in[19];
  auto BLK = [&](int l, int j) { return (const float*)d_in[20 + l * 14 + j]; };
  const float* ln_f_g = (const float*)d_in[76];
  const float* ln_f_b = (const float*)d_in[77];
  const float* dec_w1 = (const float*)d_in[78];
  const float* dec_b1 = (const float*)d_in[79];
  const float* dec_w2 = (const float*)d_in[80];
  const float* dec_b2 = (const float*)d_in[81];
  const float* dec_w3 = (const float*)d_in[82];
  const float* dec_b3 = (const float*)d_in[83];

  float* out = (float*)d_out;
  const size_t REC_OFF = 0;
  const size_t MU_OFF  = (size_t)B_ * T_ * NG_;
  const size_t LV_OFF  = MU_OFF + (size_t)B_ * D_;
  const size_t LAT_OFF = LV_OFF + (size_t)B_ * D_;

  // ---- workspace bump allocator -------------------------------------------
  char* wsb = (char*)d_ws;
  size_t off = 0;
  auto alloc = [&](size_t bytes) -> void* {
    void* p = wsb + off;
    off = (off + bytes + 255) & ~(size_t)255;
    return p;
  };
  typedef _Float16 h16;

  h16* wt_enc1 = (h16*)alloc((size_t)512 * 2016 * 2);
  h16* wt_enc2 = (h16*)alloc((size_t)256 * 512 * 2);
  h16* wt_mu   = (h16*)alloc((size_t)512 * 256 * 2);
  h16* wt_lv   = (h16*)alloc((size_t)512 * 256 * 2);
  h16* wt_sp1  = (h16*)alloc((size_t)128 * 32 * 2);
  h16* wt_sp2  = (h16*)alloc((size_t)512 * 128 * 2);
  h16 *wt_qkvg[4], *wt_wo[4], *wt_f1[4], *wt_f2[4];
  for (int l = 0; l < 4; ++l) {
    wt_qkvg[l] = (h16*)alloc((size_t)2048 * 512 * 2);
    wt_wo[l]   = (h16*)alloc((size_t)512 * 512 * 2);
    wt_f1[l]   = (h16*)alloc((size_t)1024 * 512 * 2);
    wt_f2[l]   = (h16*)alloc((size_t)512 * 1024 * 2);
  }
  h16* wt_d1 = (h16*)alloc((size_t)256 * 512 * 2);
  h16* wt_d2 = (h16*)alloc((size_t)512 * 256 * 2);
  h16* wt_d3 = (h16*)alloc((size_t)2000 * 512 * 2);

  h16*   enc16  = (h16*)alloc((size_t)B_ * 2016 * 2);
  h16*   h1_16  = (h16*)alloc((size_t)B_ * 512 * 2);
  h16*   h2_16  = (h16*)alloc((size_t)B_ * 256 * 2);
  h16*   co16   = (h16*)alloc((size_t)B_ * 32 * 2);
  float* sbuf   = (float*)alloc((size_t)B_ * 128 * 4);
  h16*   s16    = (h16*)alloc((size_t)B_ * 128 * 2);
  float* zspace = (float*)alloc((size_t)B_ * 512 * 4);
  float* xbuf   = (float*)alloc((size_t)B_ * 512 * 4);
  float* x2buf  = (float*)alloc((size_t)B_ * 512 * 4);
  h16*   xn16   = (h16*)alloc((size_t)B_ * 512 * 2);
  float* qkvg   = (float*)alloc((size_t)B_ * 2048 * 4);
  h16*   gated16 = (h16*)alloc((size_t)B_ * 512 * 2);
  h16*   ffn16  = (h16*)alloc((size_t)B_ * 1024 * 2);
  h16*   cur16  = (h16*)alloc((size_t)B_ * 512 * 2);
  h16*   d1_16  = (h16*)alloc((size_t)B_ * 256 * 2);
  h16*   d2_16  = (h16*)alloc((size_t)B_ * 512 * 2);
  const size_t S_PER_L = (size_t)B_ * H_ * DH_ * DH_;  // floats
  float* Sbuf = (float*)alloc(4 * S_PER_L * 4);

  // ---- launch helpers ------------------------------------------------------
  auto conv = [&](const float* W, h16* Wt, int K, int N, int Kpad) {
    const size_t total = (size_t)N * Kpad;
    convert_wt<<<(unsigned)((total + 255) / 256), 256, 0, stream>>>(W, Wt, K, N, Kpad);
  };
  auto gemm = [&](const h16* A, const h16* Wt, int M, int N, int K,
                  const float* bias, const float* scale, float sconst,
                  const float* bias2, const float* resid,
                  float* o32, int ldc, h16* o16, int act) {
    dim3 blk(32, 4);
    if (N % 64 == 0) {
      dim3 grd(N / 64, M / 64);
      gemm_wmma<4><<<grd, blk, 0, stream>>>(A, Wt, bias, scale, sconst, bias2,
                                            resid, o32, ldc, o16, M, N, K, act);
    } else {
      dim3 grd(N / 16, M / 64);
      gemm_wmma<1><<<grd, blk, 0, stream>>>(A, Wt, bias, scale, sconst, bias2,
                                            resid, o32, ldc, o16, M, N, K, act);
    }
  };

  // ---- weight conversion (f32 -> f16, transposed [N,Kpad]) -----------------
  conv(enc_w1, wt_enc1, 2002, 512, 2016);
  conv(enc_w2, wt_enc2, 512, 256, 512);
  conv(mu_w,   wt_mu,   256, 512, 256);
  conv(lv_w,   wt_lv,   256, 512, 256);
  conv(sp_w1,  wt_sp1,  2, 128, 32);
  conv(sp_w2,  wt_sp2,  128, 512, 128);
  for (int l = 0; l < 4; ++l) {
    conv(BLK(l, 2), wt_qkvg[l] + (size_t)0 * 512 * 512, 512, 512, 512);  // wq
    conv(BLK(l, 3), wt_qkvg[l] + (size_t)1 * 512 * 512, 512, 512, 512);  // wk
    conv(BLK(l, 4), wt_qkvg[l] + (size_t)2 * 512 * 512, 512, 512, 512);  // wv
    conv(BLK(l, 5), wt_qkvg[l] + (size_t)3 * 512 * 512, 512, 512, 512);  // wg
    conv(BLK(l, 6), wt_wo[l], 512, 512, 512);
    conv(BLK(l, 10), wt_f1[l], 512, 1024, 512);
    conv(BLK(l, 12), wt_f2[l], 1024, 512, 1024);
  }
  conv(dec_w1, wt_d1, 512, 256, 512);
  conv(dec_w2, wt_d2, 256, 512, 256);
  conv(dec_w3, wt_d3, 512, 2000, 512);

  // zero retention states
  hipMemsetAsync(Sbuf, 0, 4 * S_PER_L * 4, stream);

  // ---- encoder -------------------------------------------------------------
  const float bnsc = 1.0f / sqrtf(1.0f + 1e-5f);
  build_enc<<<(unsigned)(((size_t)B_ * 2016 + 255) / 256), 256, 0, stream>>>(
      x_genes, x_coords, enc16);
  gemm(enc16, wt_enc1, B_, 512, 2016, enc_b1, bn1_g, bnsc, bn1_b, nullptr,
       nullptr, 0, h1_16, 1);
  gemm(h1_16, wt_enc2, B_, 256, 512, enc_b2, bn2_g, bnsc, bn2_b, nullptr,
       nullptr, 0, h2_16, 1);
  gemm(h2_16, wt_mu, B_, 512, 256, mu_b, nullptr, 1.f, nullptr, nullptr,
       out + MU_OFF, 512, nullptr, 0);
  gemm(h2_16, wt_lv, B_, 512, 256, lv_b, nullptr, 1.f, nullptr, nullptr,
       out + LV_OFF, 512, nullptr, 0);

  build_coords<<<(B_ * 32 + 255) / 256, 256, 0, stream>>>(x_coords, co16);
  gemm(co16, wt_sp1, B_, 128, 32, sp_b1, nullptr, 1.f, nullptr, nullptr,
       sbuf, 128, nullptr, 0);
  ln_kernel<<<B_, 128, 0, stream>>>(sbuf, sp_ln_g, sp_ln_b, s16,
                                    nullptr, 0, nullptr, 0, 128, 1);
  gemm(s16, wt_sp2, B_, 512, 128, sp_b2, nullptr, 1.f, nullptr, nullptr,
       zspace, 512, nullptr, 0);
  add2<<<(B_ * D_ + 255) / 256, 256, 0, stream>>>(out + MU_OFF, zspace, xbuf, B_ * D_);

  // ---- recurrent decode loop ----------------------------------------------
  for (int t = 0; t < T_; ++t) {
    for (int l = 0; l < 4; ++l) {
      ln_kernel<<<B_, 128, 0, stream>>>(xbuf, BLK(l, 0), BLK(l, 1), xn16,
                                        nullptr, 0, nullptr, 0, D_, 0);
      gemm(xn16, wt_qkvg[l], B_, 2048, 512, nullptr, nullptr, 1.f, nullptr,
           nullptr, qkvg, 2048, nullptr, 0);
      retention_kernel<<<B_ * H_, 64, 0, stream>>>(
          qkvg, Sbuf + (size_t)l * S_PER_L, BLK(l, 7), gated16);
      gemm(gated16, wt_wo[l], B_, 512, 512, nullptr, nullptr, 1.f, nullptr,
           xbuf, x2buf, 512, nullptr, 0);
      ln_kernel<<<B_, 128, 0, stream>>>(x2buf, BLK(l, 8), BLK(l, 9), xn16,
                                        nullptr, 0, nullptr, 0, D_, 0);
      gemm(xn16, wt_f1[l], B_, 1024, 512, BLK(l, 11), nullptr, 1.f, nullptr,
           nullptr, nullptr, 0, ffn16, 2);
      gemm(ffn16, wt_f2[l], B_, 512, 1024, BLK(l, 13), nullptr, 1.f, nullptr,
           x2buf, xbuf, 512, nullptr, 0);
    }
    // final LN: next-step token input (in place), latent output, f16 for dec
    ln_kernel<<<B_, 128, 0, stream>>>(xbuf, ln_f_g, ln_f_b, cur16,
                                      xbuf, D_, out + LAT_OFF + (size_t)t * D_,
                                      T_ * D_, D_, 0);
    // decoder
    gemm(cur16, wt_d1, B_, 256, 512, dec_b1, nullptr, 1.f, nullptr, nullptr,
         nullptr, 0, d1_16, 1);
    gemm(d1_16, wt_d2, B_, 512, 256, dec_b2, nullptr, 1.f, nullptr, nullptr,
         nullptr, 0, d2_16, 1);
    gemm(d2_16, wt_d3, B_, 2000, 512, dec_b3, nullptr, 1.f, nullptr, nullptr,
         out + REC_OFF + (size_t)t * NG_, T_ * NG_, nullptr, 0);
  }
}
